// DynamicalNetwork_86208583565970
// MI455X (gfx1250) — compile-verified, matrix operand order checked by reading, not run on hardware
//
#include <hip/hip_runtime.h>
#include <hip/hip_bf16.h>

#define B_   16
#define T_   256
#define IN_  64
#define N_   512
#define TR_  32
#define OUT_ 64
#define LDK_ 520   // padded LDS row stride (elements); 1040B = 65*16B, bank-spread + 16B aligned

typedef __attribute__((ext_vector_type(16))) __bf16 v16bf;
typedef __attribute__((ext_vector_type(8)))  __bf16 v8bf;
typedef __attribute__((ext_vector_type(4)))  __bf16 v4bf;
typedef __attribute__((ext_vector_type(8)))  float  v8f;
typedef __attribute__((ext_vector_type(4)))  float  v4f;

// Fragment loader for both A (row-major activations [M][K]) and B (row-major
// weights [Ncol][K], i.e. X @ W^T form). CDNA5 16-bit A layout (wave32):
//   lane<16 : row=lane,    K = {kb..kb+7, kb+16..kb+23} with kb=0
//   lane>=16: row=lane-16, K same chunks with kb=8
// Two 16-byte LDS/global loads per lane.
__device__ __forceinline__ v16bf load_frag(const __bf16* __restrict__ base, int ldk, int lane) {
  const int r  = lane & 15;
  const int kb = (lane >> 4) << 3;
  const __bf16* p = base + r * ldk + kb;
  v8bf lo = *(const v8bf*)(p);
  v8bf hi = *(const v8bf*)(p + 16);
  v16bf f;
#pragma unroll
  for (int i = 0; i < 8; ++i) { f[i] = lo[i]; f[i + 8] = hi[i]; }
  return f;
}

__device__ __forceinline__ v8f wmma_bf16(v16bf a, v16bf b, v8f c) {
  return __builtin_amdgcn_wmma_f32_16x16x32_bf16(false, a, false, b, (short)0, c, false, false);
}

// ---------------------------------------------------------------- converts
__global__ void cvt_f32_bf16(const float* __restrict__ s, __bf16* __restrict__ d, int n) {
  int i = blockIdx.x * blockDim.x + threadIdx.x;
  int stride = gridDim.x * blockDim.x;
  for (; i < n; i += stride) d[i] = (__bf16)s[i];
}

// ---------------------------------------------------------------- Cholesky
// One workgroup, 256 threads. L (f32, lower-tri, upper zeroed) into ws, then
// Ls = L * sqrt(dt) in bf16 (sqrt(dt) folded so the eps GEMM yields dxi).
__global__ void chol_kernel(const float* __restrict__ Sig, float* __restrict__ L,
                            __bf16* __restrict__ Ls, const float* __restrict__ dtp) {
  const int tid = threadIdx.x;
  __shared__ float red[256];
  __shared__ float sDiag;
  for (int i = tid; i < N_ * N_; i += 256) L[i] = 0.f;
  __syncthreads();
  for (int j = 0; j < N_; ++j) {
    // diagonal: parallel partial dots + tree reduce
    float s = 0.f;
    for (int k = tid; k < j; k += 256) { float v = L[j * N_ + k]; s += v * v; }
    red[tid] = s; __syncthreads();
    for (int off = 128; off > 0; off >>= 1) {
      if (tid < off) red[tid] += red[tid + off];
      __syncthreads();
    }
    if (tid == 0) { float d = sqrtf(Sig[j * N_ + j] - red[0]); sDiag = d; L[j * N_ + j] = d; }
    __syncthreads();
    const float dinv = 1.f / sDiag;
    for (int i = j + 1 + tid; i < N_; i += 256) {
      float acc = Sig[i * N_ + j];
      for (int k = 0; k < j; ++k) acc -= L[i * N_ + k] * L[j * N_ + k];
      L[i * N_ + j] = acc * dinv;
    }
    __syncthreads();
  }
  const float sdt = sqrtf(dtp[0]);
  for (int i = tid; i < N_ * N_; i += 256) Ls[i] = (__bf16)(L[i] * sdt);
}

// ---------------------------------------------------------------- h_neural
// h_neural[4096,512] = h_seq[4096,64] @ in_w[512,64]^T + in_b. One 16x16 tile
// per wave, K=64 -> 2 WMMA. 1024 blocks * 8 waves = 8192 tiles.
__global__ void __launch_bounds__(256) hneural_kernel(
    const __bf16* __restrict__ hs, const __bf16* __restrict__ iw,
    const float* __restrict__ inb, float* __restrict__ hN) {
  const int lane = threadIdx.x & 31;
  const int wid  = threadIdx.x >> 5;
  const int wt   = blockIdx.x * 8 + wid;
  const int rt   = wt >> 5;   // 0..255 row tile
  const int ct   = wt & 31;   // 0..31  col tile
  v8f acc = {};
#pragma unroll
  for (int kk = 0; kk < 2; ++kk) {
    v16bf a = load_frag(hs + (size_t)rt * 16 * IN_ + kk * 32, IN_, lane);
    v16bf b = load_frag(iw + (size_t)ct * 16 * IN_ + kk * 32, IN_, lane);
    acc = wmma_bf16(a, b, acc);
  }
  const int col = ct * 16 + (lane & 15);
  const int rb  = (lane >> 4) << 3;
  const float bias = inb[col];
#pragma unroll
  for (int g = 0; g < 8; ++g)
    hN[(size_t)(rt * 16 + rb + g) * N_ + col] = acc[g] + bias;
}

// ---------------------------------------------------------------- RNN scan
// Persistent: 32 WGs, each owns 16 rows of U (one batch b = blockIdx>>1) for
// all 256 steps. u kept in registers in C/D fragment layout; f(u) bf16
// double-buffered in LDS; eps staged f32->bf16 each step.
__global__ void __launch_bounds__(256) rnn_kernel(
    const __bf16* __restrict__ Wb,  const __bf16* __restrict__ Lsb,
    const __bf16* __restrict__ owb, const float* __restrict__ hN,
    const float* __restrict__ outb, const float* __restrict__ u0,
    const float* __restrict__ noise, float* __restrict__ out,
    const float* __restrict__ dtp, const float* __restrict__ taup,
    const float* __restrict__ kp) {
  const int tid  = threadIdx.x;
  const int lane = tid & 31;
  const int wid  = tid >> 5;        // 8 waves
  const int tile = blockIdx.x;      // 0..31
  const int b    = tile >> 1;       // TR=32 -> 2 row tiles per batch
  const int row0 = tile * 16;       // global row base (b*TR+p flattening)

  __shared__ __attribute__((aligned(16))) __bf16 r_lds[2][16][LDK_]; // f(u), double buffered
  __shared__ __attribute__((aligned(16))) __bf16 e_lds[16][LDK_];    // eps_t (bf16)
  __shared__ __attribute__((aligned(16))) float  h_lds[N_];
  __shared__ float ob_lds[OUT_];

  const float dt = dtp[0], tau = taup[0], kgain = kp[0];
  const float S_ = dt / tau;        // dt/tau
  const float A_ = 1.f - S_;        // 1 - dt/tau

  if (tid < OUT_) ob_lds[tid] = outb[tid];

  const int col15 = lane & 15;
  const int rb    = (lane >> 4) << 3;

  // u0 -> registers (fragment layout), f(u0) -> r_lds[0]
  v8f ur[4];
#pragma unroll
  for (int c = 0; c < 4; ++c) {
    const int ct  = wid + c * 8;
    const int col = ct * 16 + col15;
#pragma unroll
    for (int g = 0; g < 8; ++g) {
      float v = u0[(size_t)(row0 + rb + g) * N_ + col];
      ur[c][g] = v;
      float rv = v > 0.f ? kgain * v * v : 0.f;
      r_lds[0][rb + g][col] = (__bf16)rv;
    }
  }
  __syncthreads();

  for (int t = 0; t < T_; ++t) {
    const int cur = t & 1, nxt = cur ^ 1;

    // ---- stage eps_t (f32 -> bf16) and h_t into LDS
    {
      const int row = tid >> 4;         // 0..15
      const int k0  = (tid & 15) * 32;  // 0..480
      const int gr  = row0 + row;
      const float* ep = noise + ((size_t)((t * B_ + (gr >> 5)) * TR_ + (gr & 31))) * N_ + k0;
#pragma unroll
      for (int q = 0; q < 8; ++q) {
        v4f v = *(const v4f*)(ep + q * 4);
        v4bf bv;
#pragma unroll
        for (int e = 0; e < 4; ++e) bv[e] = (__bf16)v[e];
        *(v4bf*)(&e_lds[row][k0 + q * 4]) = bv;
      }
      if (tid < 128) {
        v4f hv = *(const v4f*)(hN + (size_t)(b * T_ + t) * N_ + tid * 4);
        *(v4f*)(&h_lds[tid * 4]) = hv;
      }
    }
    __syncthreads();

    // ---- recurrent + noise GEMMs; Euler update in registers; f(u_new) -> r_lds[nxt]
#pragma unroll
    for (int c = 0; c < 4; ++c) {
      const int ct = wid + c * 8;
      const __bf16* wbase = Wb  + (size_t)ct * 16 * N_;
      const __bf16* lbase = Lsb + (size_t)ct * 16 * N_;
      v8f aw = {};
      v8f ae = {};
      for (int k = 0; k < 16; ++k) {
        v16bf ar = load_frag(&r_lds[cur][0][0] + k * 32, LDK_, lane);
        v16bf az = load_frag(&e_lds[0][0]      + k * 32, LDK_, lane);
        v16bf bw = load_frag(wbase + k * 32, N_, lane);
        v16bf bl = load_frag(lbase + k * 32, N_, lane);
        aw = wmma_bf16(ar, bw, aw);   // f(u) @ W^T
        ae = wmma_bf16(az, bl, ae);   // eps @ (L*sqrt_dt)^T == dxi
      }
      const int col = ct * 16 + col15;
      const float hv = h_lds[col];
#pragma unroll
      for (int g = 0; g < 8; ++g) {
        float un = A_ * ur[c][g] + S_ * (aw[g] + hv) + ae[g];
        ur[c][g] = un;
        float rv = un > 0.f ? kgain * un * un : 0.f;
        r_lds[nxt][rb + g][col] = (__bf16)rv;
      }
    }
    __syncthreads();

    // ---- readout: out_t = f(u_new) @ out_w^T + out_b (waves 0..3, OUT=64)
    if (wid < 4) {
      const __bf16* obase = owb + (size_t)wid * 16 * N_;
      v8f ao = {};
      for (int k = 0; k < 16; ++k) {
        v16bf a  = load_frag(&r_lds[nxt][0][0] + k * 32, LDK_, lane);
        v16bf bo = load_frag(obase + k * 32, N_, lane);
        ao = wmma_bf16(a, bo, ao);
      }
      const int o  = wid * 16 + col15;
      const float bb = ob_lds[o];
#pragma unroll
      for (int g = 0; g < 8; ++g)
        out[((size_t)(row0 + rb + g) * T_ + t) * OUT_ + o] = ao[g] + bb;
    }
    // no barrier needed: next stage writes e_lds/h_lds (disjoint from r_lds),
    // and the pre-GEMM barrier orders everything else.
  }
}

// ---------------------------------------------------------------- launcher
extern "C" void kernel_launch(void* const* d_in, const int* in_sizes, int n_in,
                              void* d_out, int out_size, void* d_ws, size_t ws_size,
                              hipStream_t stream) {
  (void)in_sizes; (void)n_in; (void)out_size; (void)ws_size;

  const float* h_seq = (const float*)d_in[0];   // [B,T,IN]
  const float* Sig   = (const float*)d_in[1];   // [N,N]
  const float* W     = (const float*)d_in[2];   // [N,N]
  const float* in_w  = (const float*)d_in[3];   // [N,IN]
  const float* in_b  = (const float*)d_in[4];   // [N]
  const float* out_w = (const float*)d_in[5];   // [OUT,N]
  const float* out_b = (const float*)d_in[6];   // [OUT]
  const float* u0    = (const float*)d_in[7];   // [B,TR,N]
  const float* noise = (const float*)d_in[8];   // [T,B,TR,N]
  const float* dt    = (const float*)d_in[9];
  const float* tau   = (const float*)d_in[10];
  const float* kk    = (const float*)d_in[11];
  // d_in[12] = n (==2, hardcoded as u*u)

  char* ws = (char*)d_ws;
  float*  L    = (float*)ws;  ws += (size_t)N_ * N_ * 4;        // 1 MB
  __bf16* Wb   = (__bf16*)ws; ws += (size_t)N_ * N_ * 2;        // 512 KB
  __bf16* Lsb  = (__bf16*)ws; ws += (size_t)N_ * N_ * 2;        // 512 KB
  __bf16* owb  = (__bf16*)ws; ws += (size_t)OUT_ * N_ * 2;      // 64 KB
  __bf16* inwb = (__bf16*)ws; ws += (size_t)N_ * IN_ * 2;       // 64 KB
  __bf16* hsb  = (__bf16*)ws; ws += (size_t)B_ * T_ * IN_ * 2;  // 512 KB
  float*  hN   = (float*)ws;  ws += (size_t)B_ * T_ * N_ * 4;   // 8 MB

  cvt_f32_bf16<<<256, 256, 0, stream>>>(W,     Wb,   N_ * N_);
  cvt_f32_bf16<<<64,  256, 0, stream>>>(out_w, owb,  OUT_ * N_);
  cvt_f32_bf16<<<64,  256, 0, stream>>>(in_w,  inwb, N_ * IN_);
  cvt_f32_bf16<<<256, 256, 0, stream>>>(h_seq, hsb,  B_ * T_ * IN_);

  chol_kernel<<<1, 256, 0, stream>>>(Sig, L, Lsb, dt);

  hneural_kernel<<<1024, 256, 0, stream>>>(hsb, inwb, in_b, hN);

  rnn_kernel<<<32, 256, 0, stream>>>(Wb, Lsb, owb, hN, out_b, u0, noise,
                                     (float*)d_out, dt, tau, kk);
}
